// SwitchMoeWrapper_9053791060362
// MI455X (gfx1250) — matrix operand order repaired and checked by reading, not compiled, not run on hardware
//
#include <hip/hip_runtime.h>
#include <hip/hip_bf16.h>

// Problem constants (from reference): B=8, S=2048, D=2048, F=8192, E=8
constexpr int Dc = 2048;
constexpr int Fc = 8192;
constexpr int Ec = 8;
constexpr int BS = 16384;   // B*S tokens

// GEMM tiling
constexpr int BM = 128;     // token tile
constexpr int BN = 128;     // output-col tile
constexpr int BK = 32;      // K step == WMMA bf16 K
constexpr int LDA = 40;     // LDS row stride in halfwords (32 + 8 pad; 80B = 5*16B)

typedef __attribute__((ext_vector_type(16))) __bf16 v16bf;
typedef __attribute__((ext_vector_type(8)))  float  v8f;
typedef __attribute__((ext_vector_type(4)))  unsigned int v4u;
typedef __attribute__((ext_vector_type(8)))  int    v8i;
typedef __attribute__((ext_vector_type(4)))  int    v4i;

union FragU {
  v16bf v;
  uint4 q[2];
};

__device__ __forceinline__ unsigned short f2bf(float f) {
  unsigned u = __builtin_bit_cast(unsigned, f);
  u += 0x7fffu + ((u >> 16) & 1u);      // round-to-nearest-even
  return (unsigned short)(u >> 16);
}

// A fragment (16x32 bf16, M x K): lane L holds row M=L&15;
// halves 0..7  = K (L<16 ? 0 : 8) + 0..7,  halves 8..15 = 16 + same  -> two b128 LDS loads
__device__ __forceinline__ v16bf load_a_frag(const unsigned short* As, int m, int lane) {
  const int kg = (lane >> 4) << 3;   // 0 or 8
  FragU f;
  f.q[0] = *(const uint4*)(As + m * LDA + kg);
  f.q[1] = *(const uint4*)(As + m * LDA + 16 + kg);
  return f.v;
}

// B fragment (32x16 bf16, K x N), staged transposed in LDS as Bt[n][k]:
// lane L holds col N=L&15; halves 0..7 = K (L<16 ? 0 : 8) + 0..7, halves 8..15 = 16 + same -> two b128 LDS loads
__device__ __forceinline__ v16bf load_b_frag(const unsigned short* Bt, int n, int lane) {
  const int kg = (lane >> 4) << 3;   // 0 or 8
  FragU f;
  f.q[0] = *(const uint4*)(Bt + n * LDA + kg);
  f.q[1] = *(const uint4*)(Bt + n * LDA + 16 + kg);
  return f.v;
}

// ---------------------------------------------------------------- init / scan
__global__ void moe_init(int* __restrict__ counts) {
  if (threadIdx.x < Ec) counts[threadIdx.x] = 0;
}

__global__ void moe_scan(const int* __restrict__ counts, int* __restrict__ bases) {
  if (threadIdx.x == 0) {
    int a = 0;
    for (int e = 0; e < Ec; ++e) { bases[e] = a; a += counts[e]; }
  }
}

// ---------------------------------------------------------------- router
__global__ __launch_bounds__(256) void moe_router(
    const float* __restrict__ X, const float* __restrict__ Wg,
    float* __restrict__ logits, int* __restrict__ eidx,
    float* __restrict__ probs, int* __restrict__ counts, int* __restrict__ tlist) {
  const int wid  = threadIdx.x >> 5;
  const int lane = threadIdx.x & 31;
  const int t    = blockIdx.x * 8 + wid;

  float acc[Ec];
#pragma unroll
  for (int e = 0; e < Ec; ++e) acc[e] = 0.f;

  const float* xr = X + (size_t)t * Dc;
  for (int d = lane; d < Dc; d += 32) {
    const float  x  = xr[d];
    const float4 w0 = *(const float4*)(Wg + (size_t)d * Ec);
    const float4 w1 = *(const float4*)(Wg + (size_t)d * Ec + 4);
    acc[0] += x * w0.x; acc[1] += x * w0.y; acc[2] += x * w0.z; acc[3] += x * w0.w;
    acc[4] += x * w1.x; acc[5] += x * w1.y; acc[6] += x * w1.z; acc[7] += x * w1.w;
  }
#pragma unroll
  for (int e = 0; e < Ec; ++e) {
#pragma unroll
    for (int off = 16; off > 0; off >>= 1)
      acc[e] += __shfl_xor(acc[e], off, 32);
  }

  if (lane == 0) {
    float m = acc[0]; int best = 0;
#pragma unroll
    for (int e = 1; e < Ec; ++e) if (acc[e] > m) { m = acc[e]; best = e; }
    float s = 0.f;
#pragma unroll
    for (int e = 0; e < Ec; ++e) s += __expf(acc[e] - m);
#pragma unroll
    for (int e = 0; e < Ec; ++e) logits[(size_t)t * Ec + e] = acc[e];
    eidx[t]  = best;
    probs[t] = 1.0f / s;
    const int pos = atomicAdd(&counts[best], 1);
    tlist[best * BS + pos] = t;
  }
}

// ---------------------------------------------------------------- GEMM1
// H[base_e + slot, :] = relu( X[token(slot), :] @ wi[e] )   (bf16 out)
__global__ __launch_bounds__(256) void moe_gemm1(
    const float* __restrict__ X, const float* __restrict__ Wi_all,
    const int* __restrict__ tlist, const int* __restrict__ counts,
    const int* __restrict__ bases, unsigned short* __restrict__ H) {
  const int e   = blockIdx.z;
  const int cnt = counts[e];
  const int tm  = blockIdx.y;
  if (tm * BM >= cnt) return;
  const int f0   = blockIdx.x * BN;
  const int base = bases[e];
  const float* Wi = Wi_all + (size_t)e * Dc * Fc;

  __shared__ unsigned short As[BM * LDA];
  __shared__ unsigned short Bt[BN * LDA];
  __shared__ int s_tok[BM];

  const int tid  = threadIdx.x;
  const int lane = tid & 31;
  const int wid  = tid >> 5;
  const int wm   = wid >> 2;   // 0..1  (64 rows each)
  const int wn   = wid & 3;    // 0..3  (32 cols each)

  if (tid < BM) {
    int slot = tm * BM + tid;
    if (slot >= cnt) slot = cnt - 1;
    s_tok[tid] = tlist[e * BS + slot];
  }
  __syncthreads();

  v8f acc[4][2] = {};

  for (int k0 = 0; k0 < Dc; k0 += BK) {
    // stage A: gathered hidden rows, f32 -> bf16, row-major [128][32]
#pragma unroll
    for (int it = 0; it < 4; ++it) {
      const int j  = tid + 256 * it;      // 0..1023 float4s
      const int r  = j >> 3;
      const int c4 = j & 7;
      const float4 xv = *(const float4*)(X + (size_t)s_tok[r] * Dc + k0 + c4 * 4);
      ushort4 h;
      h.x = f2bf(xv.x); h.y = f2bf(xv.y); h.z = f2bf(xv.z); h.w = f2bf(xv.w);
      *(ushort4*)(&As[r * LDA + c4 * 4]) = h;
    }
    // stage B: wi[e][k][f] tile, f32 -> bf16, transposed to Bt[n][k]
#pragma unroll
    for (int it = 0; it < 4; ++it) {
      const int j  = tid + 256 * it;
      const int kk = j >> 5;
      const int c4 = j & 31;
      const float4 wv = *(const float4*)(Wi + (size_t)(k0 + kk) * Fc + f0 + c4 * 4);
      Bt[(c4 * 4 + 0) * LDA + kk] = f2bf(wv.x);
      Bt[(c4 * 4 + 1) * LDA + kk] = f2bf(wv.y);
      Bt[(c4 * 4 + 2) * LDA + kk] = f2bf(wv.z);
      Bt[(c4 * 4 + 3) * LDA + kk] = f2bf(wv.w);
    }
    // speculative prefetch of next K-step's weight tile (dropped if OOB)
    __builtin_prefetch(Wi + (size_t)(k0 + BK + (tid >> 3)) * Fc + f0 + (tid & 7) * 16, 0, 0);
    __syncthreads();

    v16bf afr[4], bfr[2];
#pragma unroll
    for (int mt = 0; mt < 4; ++mt)
      afr[mt] = load_a_frag(As, wm * 64 + mt * 16 + (lane & 15), lane);
#pragma unroll
    for (int nt = 0; nt < 2; ++nt)
      bfr[nt] = load_b_frag(Bt, wn * 32 + nt * 16 + (lane & 15), lane);
#pragma unroll
    for (int mt = 0; mt < 4; ++mt)
#pragma unroll
      for (int nt = 0; nt < 2; ++nt)
        acc[mt][nt] = __builtin_amdgcn_wmma_f32_16x16x32_bf16(
            false, afr[mt], false, bfr[nt], (short)0, acc[mt][nt], false, false);
    __syncthreads();
  }

  // epilogue: relu -> bf16 -> compact H  (guard hoisted out of the hot path)
  const int l15 = lane & 15;
  const int hi8 = (lane >> 4) << 3;
  const int rbase = tm * BM + wm * 64 + hi8;
  const int cbase = f0 + wn * 32 + l15;
  if (tm * BM + BM <= cnt) {
#pragma unroll
    for (int mt = 0; mt < 4; ++mt)
#pragma unroll
      for (int nt = 0; nt < 2; ++nt)
#pragma unroll
        for (int v = 0; v < 8; ++v) {
          float x = acc[mt][nt][v];
          x = x > 0.f ? x : 0.f;
          H[(size_t)(base + rbase + mt * 16 + v) * Fc + cbase + nt * 16] = f2bf(x);
        }
  } else {
#pragma unroll
    for (int mt = 0; mt < 4; ++mt)
#pragma unroll
      for (int nt = 0; nt < 2; ++nt)
#pragma unroll
        for (int v = 0; v < 8; ++v) {
          const int slot = rbase + mt * 16 + v;
          if (slot < cnt) {
            float x = acc[mt][nt][v];
            x = x > 0.f ? x : 0.f;
            H[(size_t)(base + slot) * Fc + cbase + nt * 16] = f2bf(x);
          }
        }
  }
}

// ---------------------------------------------------------------- GEMM2
// out[token(slot), :] = prob(token) * ( H[base_e + slot, :] @ wo[e] )
// A-operand (H, bf16, fixed row stride) is staged LDS-side by the Tensor Data
// Mover: one tensor_load_to_lds per K-step; D# pad fields reproduce the LDA=40
// LDS row stride and tensor_dim1 zero-fills rows past the expert's token count.
__global__ __launch_bounds__(256) void moe_gemm2(
    const unsigned short* __restrict__ H, const float* __restrict__ Wo_all,
    const int* __restrict__ tlist, const int* __restrict__ counts,
    const int* __restrict__ bases, const float* __restrict__ probs,
    float* __restrict__ out) {
  const int e   = blockIdx.z;
  const int cnt = counts[e];
  const int tm  = blockIdx.y;
  if (tm * BM >= cnt) return;
  const int d0   = blockIdx.x * BN;
  const int base = bases[e];
  const float* Wo = Wo_all + (size_t)e * Fc * Dc;

  __shared__ unsigned short As[BM * LDA];
  __shared__ unsigned short Bt[BN * LDA];
  __shared__ int   s_tok[BM];
  __shared__ float s_prob[BM];

  const int tid  = threadIdx.x;
  const int lane = tid & 31;
  const int wid  = tid >> 5;
  const int wm   = wid >> 2;
  const int wn   = wid & 3;

  if (tid < BM) {
    int slot = tm * BM + tid;
    if (slot >= cnt) slot = cnt - 1;
    const int t = tlist[e * BS + slot];
    s_tok[tid]  = t;
    s_prob[tid] = probs[t];
  }
  __syncthreads();

  // TDM descriptor constants (wave-uniform)
  const int rem = cnt - tm * BM;                           // rows valid in this tile
  const unsigned ldsOff = (unsigned)(uintptr_t)(&As[0]);   // LDS byte offset
  const unsigned long long gaRow =
      (unsigned long long)(uintptr_t)(H + (size_t)(base + tm * BM) * Fc);
  // group1: wg_mask=0 | data_size=2B | pad_enable | pad_interval=16dw | pad_amount=4dw
  const int g1d0 = (1 << 16) | (1 << 20) | (3 << 22) | (3 << 25);

  v8f acc[4][2] = {};

  for (int k0 = 0; k0 < Fc; k0 += BK) {
    // A tile via Tensor Data Mover (wave 0 issues; TENSORcnt waited before barrier)
    if (tid < 32) {
      const unsigned long long ga = gaRow + (unsigned long long)k0 * 2ull;
      v4u g0 = { 1u,                                    // count=1 valid descriptor
                 ldsOff,                                // lds_addr
                 (unsigned)ga,                          // global_addr[31:0]
                 (unsigned)(ga >> 32) | (2u << 30) };   // global_addr[56:32] | type=2
      v8i g1 = { g1d0,
                 (int)((Fc & 0xffff) << 16),            // tensor_dim0[15:0] @bit48
                 (int)(((unsigned)rem & 0xffffu) << 16),// dim0 hi=0 | tensor_dim1[15:0]
                 (int)(32u << 16),                      // dim1 hi=0 | tile_dim0=32
                 128,                                   // tile_dim1=128 | tile_dim2=0
                 Fc,                                    // tensor_dim0_stride lo32
                 0, 0 };                                // stride hi | dim1_stride
      v4i gz  = { 0, 0, 0, 0 };
      v8i gz8 = { 0, 0, 0, 0, 0, 0, 0, 0 };
      __builtin_amdgcn_tensor_load_to_lds(g0, g1, gz, gz, gz8, 0);
    }
    // stage B: wo[e][k][d] tile, f32 -> bf16, transposed to Bt[n][k]
#pragma unroll
    for (int it = 0; it < 4; ++it) {
      const int j  = tid + 256 * it;
      const int kk = j >> 5;
      const int c4 = j & 31;
      const float4 wv = *(const float4*)(Wo + (size_t)(k0 + kk) * Dc + d0 + c4 * 4);
      Bt[(c4 * 4 + 0) * LDA + kk] = f2bf(wv.x);
      Bt[(c4 * 4 + 1) * LDA + kk] = f2bf(wv.y);
      Bt[(c4 * 4 + 2) * LDA + kk] = f2bf(wv.z);
      Bt[(c4 * 4 + 3) * LDA + kk] = f2bf(wv.w);
    }
    __builtin_prefetch(Wo + (size_t)(k0 + BK + (tid >> 3)) * Dc + d0 + (tid & 7) * 16, 0, 0);
    if (tid < 32) __builtin_amdgcn_s_wait_tensorcnt(0);
    __syncthreads();

    v16bf afr[4], bfr[2];
#pragma unroll
    for (int mt = 0; mt < 4; ++mt)
      afr[mt] = load_a_frag(As, wm * 64 + mt * 16 + (lane & 15), lane);
#pragma unroll
    for (int nt = 0; nt < 2; ++nt)
      bfr[nt] = load_b_frag(Bt, wn * 32 + nt * 16 + (lane & 15), lane);
#pragma unroll
    for (int mt = 0; mt < 4; ++mt)
#pragma unroll
      for (int nt = 0; nt < 2; ++nt)
        acc[mt][nt] = __builtin_amdgcn_wmma_f32_16x16x32_bf16(
            false, afr[mt], false, bfr[nt], (short)0, acc[mt][nt], false, false);
    __syncthreads();
  }

  // epilogue: scale by routing prob, scatter to the token's output row
  const int l15 = lane & 15;
  const int hi8 = (lane >> 4) << 3;
  const int rloc = wm * 64 + hi8;
  const int cbase = d0 + wn * 32 + l15;
  if (tm * BM + BM <= cnt) {
#pragma unroll
    for (int mt = 0; mt < 4; ++mt)
#pragma unroll
      for (int nt = 0; nt < 2; ++nt)
#pragma unroll
        for (int v = 0; v < 8; ++v) {
          const int r = rloc + mt * 16 + v;
          out[(size_t)s_tok[r] * Dc + cbase + nt * 16] = acc[mt][nt][v] * s_prob[r];
        }
  } else {
#pragma unroll
    for (int mt = 0; mt < 4; ++mt)
#pragma unroll
      for (int nt = 0; nt < 2; ++nt)
#pragma unroll
        for (int v = 0; v < 8; ++v) {
          const int r = rloc + mt * 16 + v;
          if (tm * BM + r < cnt)
            out[(size_t)s_tok[r] * Dc + cbase + nt * 16] = acc[mt][nt][v] * s_prob[r];
        }
  }
}

// ---------------------------------------------------------------- launch
extern "C" void kernel_launch(void* const* d_in, const int* in_sizes, int n_in,
                              void* d_out, int out_size, void* d_ws, size_t ws_size,
                              hipStream_t stream) {
  const float* X  = (const float*)d_in[0];   // [B,S,D]
  const float* Wg = (const float*)d_in[1];   // [D,E]
  const float* Wi = (const float*)d_in[2];   // [E,D,F]
  const float* Wo = (const float*)d_in[3];   // [E,F,D]

  float* out    = (float*)d_out;                       // [B,S,D]
  float* logits = out + (size_t)BS * Dc;               // [B,S,E]
  int*   eidx   = (int*)(logits + (size_t)BS * Ec);    // [B,S] int32 bits

  char* ws = (char*)d_ws;
  int*   counts = (int*)(ws);                              // 8 ints
  int*   bases  = (int*)(ws + 64);                         // 8 ints
  int*   tlist  = (int*)(ws + 256);                        // E*BS ints (512 KB)
  float* probs  = (float*)(ws + 256 + sizeof(int) * Ec * BS);  // BS floats
  unsigned short* H = (unsigned short*)(ws + ((size_t)1 << 20)); // [16384, F] bf16 (256 MB)

  moe_init<<<1, 32, 0, stream>>>(counts);
  moe_router<<<BS / 8, 256, 0, stream>>>(X, Wg, logits, eidx, probs, counts, tlist);
  moe_scan<<<1, 32, 0, stream>>>(counts, bases);
  moe_gemm1<<<dim3(Fc / BN, BS / BM, Ec), 256, 0, stream>>>(X, Wi, tlist, counts, bases, H);
  moe_gemm2<<<dim3(Dc / BN, BS / BM, Ec), 256, 0, stream>>>(H, Wo, tlist, counts, bases, probs, out);
}
